// SigGateMultiHeadAttention_83391085019439
// MI455X (gfx1250) — matrix operand order repaired and checked by reading, not compiled, not run on hardware
//
#include <hip/hip_runtime.h>
#include <stdint.h>

// ---------------------------------------------------------------------------
// Types for CDNA5 WMMA (wave32, 16x16x32 bf16 -> f32) and TDM descriptors
// ---------------------------------------------------------------------------
typedef __attribute__((ext_vector_type(16))) __bf16 v16bf;
typedef __attribute__((ext_vector_type(8)))  __bf16 bf16x8;
typedef __attribute__((ext_vector_type(8)))  float  v8f;
typedef __attribute__((ext_vector_type(4)))  unsigned int u32x4;
typedef __attribute__((ext_vector_type(8)))  int i32x8;
typedef __attribute__((ext_vector_type(4)))  int i32x4;

#define NEG_INF (-__builtin_inff())

#define BB 2
#define SS 2048
#define DD 1024
#define HH 16
#define HD 64

#if __has_builtin(__builtin_amdgcn_tensor_load_to_lds)
#define HAS_TDM 1
#else
#define HAS_TDM 0
#endif

__device__ __forceinline__ v8f wmma_bf16(v16bf a, v16bf b, v8f c) {
  // 8 args: (neg_a, A, neg_b, B, c_mod, C, reuse_a, reuse_b)
  return __builtin_amdgcn_wmma_f32_16x16x32_bf16(false, a, false, b, (short)0, c,
                                                 false, false);
}

// A-operand 16x32 bf16 fragment from a row-major matrix (row = M, ld elems).
// ISA layout: lanes 0-15 (m=lane): K chunks [0,8)+[16,24); lanes 16-31: [8,16)+[24,32).
__device__ __forceinline__ v16bf load_frag_a(const __bf16* p, int ld, int lane) {
  const int m = lane & 15, half = lane >> 4;
  const __bf16* row = p + (size_t)m * ld + half * 8;
  bf16x8 lo = *(const bf16x8*)(row);
  bf16x8 hi = *(const bf16x8*)(row + 16);
  v16bf f;
#pragma unroll
  for (int i = 0; i < 8; ++i) { f[i] = lo[i]; f[i + 8] = hi[i]; }
  return f;
}

// B-operand 32x16 (KxN) bf16 fragment, sourced from the TRANSPOSED matrix
// Bt (N x K, row-major). Lanes 0-15 (n=lane): K [0,8)+[16,24); lanes 16-31: [8,16)+[24,32).
__device__ __forceinline__ v16bf load_frag_b(const __bf16* bt, int ld, int lane) {
  const int n = lane & 15, half = lane >> 4;
  const __bf16* row = bt + (size_t)n * ld + half * 8;
  bf16x8 lo = *(const bf16x8*)(row);
  bf16x8 hi = *(const bf16x8*)(row + 16);
  v16bf f;
#pragma unroll
  for (int i = 0; i < 8; ++i) { f[i] = lo[i]; f[i + 8] = hi[i]; }
  return f;
}

// ---------------------------------------------------------------------------
// TDM: DMA a 2D bf16 tile (rows x 32) from global into LDS (D# per ISA ch.8).
// lds_addr = LDS byte address (low 32 bits of flat shared pointer).
// ---------------------------------------------------------------------------
__device__ __forceinline__ void tdm_load_2d(unsigned int lds_addr,
                                            const __bf16* gsrc,
                                            unsigned int tile_rows,
                                            unsigned int tile_k,
                                            unsigned int tensor_d0,
                                            unsigned int tensor_d1,
                                            unsigned int stride0) {
#if HAS_TDM
  unsigned long long ga = (unsigned long long)(uintptr_t)gsrc;
  u32x4 g0;
  g0[0] = 1u;                                             // count=1, user D#
  g0[1] = lds_addr;                                       // lds_addr[31:0]
  g0[2] = (unsigned int)ga;                               // global_addr[31:0]
  g0[3] = ((unsigned int)(ga >> 32) & 0x01FFFFFFu) | (2u << 30);  // addr[56:32]|type=2
  i32x8 g1 = {};
  g1[0] = (int)(1u << 16);                                // data_size=1 (2 bytes)
  g1[1] = (int)((tensor_d0 & 0xFFFFu) << 16);             // tensor_dim0[15:0]
  g1[2] = (int)(((tensor_d0 >> 16) & 0xFFFFu) |           // tensor_dim0[31:16]
                ((tensor_d1 & 0xFFFFu) << 16));           // tensor_dim1[15:0]
  g1[3] = (int)(((tensor_d1 >> 16) & 0xFFFFu) |           // tensor_dim1[31:16]
                ((tile_k & 0xFFFFu) << 16));              // tile_dim0
  g1[4] = (int)(tile_rows & 0xFFFFu);                     // tile_dim1 (tile_dim2=0)
  g1[5] = (int)stride0;                                   // tensor_dim0_stride[31:0]
  g1[6] = 0;
  g1[7] = 0;
  i32x4 z4 = {};
#if __has_include(<hip/amd_detail/amd_gfx1250_TDM.h>)
  i32x8 z8 = {};
  __builtin_amdgcn_tensor_load_to_lds(g0, g1, z4, z4, z8, 0);   // 6-arg toolchain
#else
  __builtin_amdgcn_tensor_load_to_lds(g0, g1, z4, z4, 0);       // 5-arg (ROCm 7.2)
#endif
#else
  (void)lds_addr; (void)gsrc; (void)tile_rows; (void)tile_k;
  (void)tensor_d0; (void)tensor_d1; (void)stride0;
#endif
}

__device__ __forceinline__ void wait_tensor0() {
#if HAS_TDM
#if __has_builtin(__builtin_amdgcn_s_wait_tensorcnt)
  __builtin_amdgcn_s_wait_tensorcnt(0);
#else
  asm volatile("s_wait_tensorcnt 0x0" ::: "memory");
#endif
#endif
}

// ---------------------------------------------------------------------------
// fp32 -> bf16 convert (x), and transpose+convert (weights, D x D)
// ---------------------------------------------------------------------------
__global__ void cvt_bf16_kernel(const float* __restrict__ src,
                                __bf16* __restrict__ dst, int n) {
  int i = blockIdx.x * 256 + threadIdx.x;
  if (i < n) dst[i] = (__bf16)src[i];
}

__global__ void transpose_cvt_kernel(const float* __restrict__ W,
                                     __bf16* __restrict__ Wt) {
  __shared__ float tile[32][33];
  const int x = blockIdx.x * 32 + threadIdx.x;   // column of W
  const int y0 = blockIdx.y * 32;                // row base of W
  for (int r = threadIdx.y; r < 32; r += 8)
    tile[r][threadIdx.x] = W[(size_t)(y0 + r) * DD + x];
  __syncthreads();
  const int n0 = blockIdx.x * 32;                // Wt row base (= W column)
  for (int r = threadIdx.y; r < 32; r += 8)
    Wt[(size_t)(n0 + r) * DD + y0 + threadIdx.x] = (__bf16)tile[threadIdx.x][r];
}

// ---------------------------------------------------------------------------
// Projection GEMM: Y = A(4096xD) @ W + b, bf16 WMMA, TDM-staged LDS tiles.
// Block: 256 threads (8 waves), tile 128(M) x 64(N); wave -> 16x64 strip.
// Double-buffered LDS: A 128x32, B 64x32 per buffer (24 KB total).
// mode 0: Q -> [b][h][s][hd]   mode 1: K -> same   mode 2: V -> [b][h][hd][s]
// mode 3: gate=sigmoid -> [b][h][s][hd]            mode 4: fp32 -> dstf[r*D+c]
// ---------------------------------------------------------------------------
__global__ void gemm_proj_kernel(const __bf16* __restrict__ A,
                                 const __bf16* __restrict__ Bt,
                                 const float* __restrict__ bias,
                                 __bf16* __restrict__ dstb,
                                 float* __restrict__ dstf, int mode) {
  __shared__ __align__(128) __bf16 Asm[2][128 * 32];
  __shared__ __align__(128) __bf16 Bsm[2][64 * 32];

  const int tid = threadIdx.x;
  const int lane = tid & 31;
  const int w = tid >> 5;
  const int blockRow = blockIdx.x * 128;
  const int colBase = blockIdx.y * 64;

  v8f acc[4] = {};

  auto stage = [&](int buf, int k0) {
#if HAS_TDM
    if (w == 0) {
      tdm_load_2d((unsigned int)(uintptr_t)&Asm[buf][0],
                  A + (size_t)blockRow * DD + k0, 128, 32, DD, BB * SS, DD);
      tdm_load_2d((unsigned int)(uintptr_t)&Bsm[buf][0],
                  Bt + (size_t)colBase * DD + k0, 64, 32, DD, DD, DD);
    }
#else
    {   // cooperative copy fallback
      int ar = tid >> 1, ac = (tid & 1) * 16;   // A: 128 rows x 32 elems
      const __bf16* asrc = A + (size_t)(blockRow + ar) * DD + k0 + ac;
      *(bf16x8*)&Asm[buf][ar * 32 + ac] = *(const bf16x8*)asrc;
      *(bf16x8*)&Asm[buf][ar * 32 + ac + 8] = *(const bf16x8*)(asrc + 8);
      int br = tid >> 2, bc = (tid & 3) * 8;    // B: 64 rows x 32 elems
      *(bf16x8*)&Bsm[buf][br * 32 + bc] =
          *(const bf16x8*)(Bt + (size_t)(colBase + br) * DD + k0 + bc);
    }
#endif
  };

  stage(0, 0);
  for (int kt = 0; kt < DD / 32; ++kt) {
    const int cur = kt & 1;
#if HAS_TDM
    if (w == 0) wait_tensor0();      // DMA for 'cur' complete (wave 0 tracks it)
#endif
    __syncthreads();                 // publish LDS tile to all waves
    if (kt + 1 < DD / 32) stage(cur ^ 1, (kt + 1) * 32);

    v16bf a = load_frag_a(&Asm[cur][w * 16 * 32], 32, lane);
#pragma unroll
    for (int j = 0; j < 4; ++j) {
      v16bf b = load_frag_b(&Bsm[cur][j * 16 * 32], 32, lane);
      acc[j] = wmma_bf16(a, b, acc[j]);
    }
    __syncthreads();                 // all reads of 'cur' done before reuse
  }

  const int col = lane & 15, half = lane >> 4;
#pragma unroll
  for (int j = 0; j < 4; ++j) {
#pragma unroll
    for (int i = 0; i < 8; ++i) {
      const int r = blockRow + w * 16 + half * 8 + i;  // global row in [0, 4096)
      const int c = colBase + j * 16 + col;            // global col in [0, 1024)
      float y = acc[j][i] + bias[c];
      if (mode == 4) {
        dstf[(size_t)r * DD + c] = y;
      } else {
        const int b = r >> 11, s = r & (SS - 1);
        const int h = c >> 6, d = c & (HD - 1);
        if (mode == 3) y = 1.f / (1.f + __expf(-y));
        size_t idx;
        if (mode == 2) idx = ((size_t)(b * HH + h) * HD + d) * SS + s;   // V^T
        else           idx = ((size_t)(b * HH + h) * SS + s) * HD + d;   // Q/K/G
        dstb[idx] = (__bf16)y;
      }
    }
  }
}

// ---------------------------------------------------------------------------
// Attention pass 1: per-row softmax max & sum (online, per-lane, shfl-merged).
// Grid: (S/128, H, B); 8 waves, each owns 16 query rows.
// ---------------------------------------------------------------------------
__global__ void attn_pass1_kernel(const __bf16* __restrict__ Qb,
                                  const __bf16* __restrict__ Kb,
                                  const float* __restrict__ attn_mask,
                                  const unsigned char* __restrict__ kpm,
                                  float* __restrict__ Mbuf,
                                  float* __restrict__ Lbuf) {
  const int lane = threadIdx.x & 31;
  const int w = threadIdx.x >> 5;
  const int b = blockIdx.z, h = blockIdx.y;
  const int qbase = blockIdx.x * 128 + w * 16;
  const int col = lane & 15, half = lane >> 4;

  const __bf16* qp = Qb + ((size_t)(b * HH + h) * SS + qbase) * HD;
  const v16bf qf0 = load_frag_a(qp, HD, lane);
  const v16bf qf1 = load_frag_a(qp + 32, HD, lane);
  const __bf16* kbase = Kb + (size_t)(b * HH + h) * SS * HD;

  float M[8], S[8];
#pragma unroll
  for (int i = 0; i < 8; ++i) { M[i] = NEG_INF; S[i] = 0.f; }

  for (int k0 = 0; k0 < SS; k0 += 16) {
    __builtin_prefetch(kbase + (size_t)(k0 + 16) * HD, 0, 1);
    v16bf kf0 = load_frag_b(kbase + (size_t)k0 * HD, HD, lane);
    v16bf kf1 = load_frag_b(kbase + (size_t)k0 * HD + 32, HD, lane);
    v8f sc = {};
    sc = wmma_bf16(qf0, kf0, sc);
    sc = wmma_bf16(qf1, kf1, sc);
    const int key = k0 + col;
    const bool pad = kpm[b * SS + key] != 0;
#pragma unroll
    for (int i = 0; i < 8; ++i) {
      const int qr = qbase + half * 8 + i;
      float s = sc[i] * 0.125f + attn_mask[(size_t)qr * SS + key];
      if (pad) s = NEG_INF;
      const float mn = fmaxf(M[i], s);
      if (mn != NEG_INF) {
        float a0 = (M[i] == NEG_INF) ? 0.f : S[i] * __expf(M[i] - mn);
        float a1 = (s == NEG_INF) ? 0.f : __expf(s - mn);
        M[i] = mn; S[i] = a0 + a1;
      }
    }
  }

#pragma unroll
  for (int i = 0; i < 8; ++i) {
#pragma unroll
    for (int off = 8; off >= 1; off >>= 1) {     // merge across 16-lane group
      float Mo = __shfl_xor(M[i], off, 32);
      float So = __shfl_xor(S[i], off, 32);
      float mn = fmaxf(M[i], Mo);
      float a0 = (M[i] == NEG_INF) ? 0.f : S[i] * __expf(M[i] - mn);
      float a1 = (Mo == NEG_INF) ? 0.f : So * __expf(Mo - mn);
      M[i] = mn; S[i] = a0 + a1;
    }
    if (col == 0) {
      size_t sidx = (size_t)(b * HH + h) * SS + qbase + half * 8 + i;
      Mbuf[sidx] = M[i];
      Lbuf[sidx] = S[i];
    }
  }
}

// ---------------------------------------------------------------------------
// Attention pass 2: P = softmax(scores); avg += P/H (atomic); O = (P @ V) * gate.
// P is staged through LDS (per-wave tile) to convert C-layout -> A-layout.
// ---------------------------------------------------------------------------
__global__ void attn_pass2_kernel(const __bf16* __restrict__ Qb,
                                  const __bf16* __restrict__ Kb,
                                  const __bf16* __restrict__ Vtb,
                                  const __bf16* __restrict__ Gb,
                                  const float* __restrict__ attn_mask,
                                  const unsigned char* __restrict__ kpm,
                                  const float* __restrict__ Mbuf,
                                  const float* __restrict__ Lbuf,
                                  float* __restrict__ avg,
                                  __bf16* __restrict__ attn_outb) {
  __shared__ __align__(16) __bf16 Pb[8][16 * 32];  // per-wave 16x32 P tile

  const int lane = threadIdx.x & 31;
  const int w = threadIdx.x >> 5;
  const int b = blockIdx.z, h = blockIdx.y;
  const int qbase = blockIdx.x * 128 + w * 16;
  const int col = lane & 15, half = lane >> 4;

  const __bf16* qp = Qb + ((size_t)(b * HH + h) * SS + qbase) * HD;
  const v16bf qf0 = load_frag_a(qp, HD, lane);
  const v16bf qf1 = load_frag_a(qp + 32, HD, lane);
  const __bf16* kbase = Kb + (size_t)(b * HH + h) * SS * HD;
  const __bf16* vtbase = Vtb + (size_t)(b * HH + h) * HD * SS;

  float mrow[8], lrow[8];
#pragma unroll
  for (int i = 0; i < 8; ++i) {
    size_t sidx = (size_t)(b * HH + h) * SS + qbase + half * 8 + i;
    mrow[i] = Mbuf[sidx];
    float lv = Lbuf[sidx];
    lrow[i] = (lv > 0.f) ? 1.f / lv : 0.f;
  }

  v8f acc_o[4] = {};

  for (int k0 = 0; k0 < SS; k0 += 32) {
    __builtin_prefetch(kbase + (size_t)(k0 + 32) * HD, 0, 1);
#pragma unroll
    for (int t = 0; t < 2; ++t) {
      const int n0 = k0 + t * 16;
      v16bf kf0 = load_frag_b(kbase + (size_t)n0 * HD, HD, lane);
      v16bf kf1 = load_frag_b(kbase + (size_t)n0 * HD + 32, HD, lane);
      v8f sc = {};
      sc = wmma_bf16(qf0, kf0, sc);
      sc = wmma_bf16(qf1, kf1, sc);
      const int key = n0 + col;
      const bool pad = kpm[b * SS + key] != 0;
#pragma unroll
      for (int i = 0; i < 8; ++i) {
        const int qr = qbase + half * 8 + i;
        float s = sc[i] * 0.125f + attn_mask[(size_t)qr * SS + key];
        float p = 0.f;
        if (!pad && s != NEG_INF && mrow[i] != NEG_INF)
          p = __expf(s - mrow[i]) * lrow[i];
        atomicAdd(&avg[((size_t)b * SS + qr) * SS + key], p * (1.f / HH));
        Pb[w][(half * 8 + i) * 32 + t * 16 + col] = (__bf16)p;
      }
    }
    // CDNA5 split counter: ensure all LDS stores landed before cross-lane reads.
    asm volatile("s_wait_dscnt 0" ::: "memory");
    v16bf pf = load_frag_a(&Pb[w][0], 32, lane);
#pragma unroll
    for (int j = 0; j < 4; ++j) {
      v16bf vf = load_frag_b(vtbase + (size_t)(j * 16) * SS + k0, SS, lane);
      acc_o[j] = wmma_bf16(pf, vf, acc_o[j]);
    }
  }

#pragma unroll
  for (int j = 0; j < 4; ++j) {
#pragma unroll
    for (int i = 0; i < 8; ++i) {
      const int qr = qbase + half * 8 + i;
      const int d = j * 16 + col;
      float g = (float)Gb[((size_t)(b * HH + h) * SS + qr) * HD + d];
      float o = acc_o[j][i] * g;
      attn_outb[((size_t)b * SS + qr) * DD + h * HD + d] = (__bf16)o;
    }
  }
}

// ---------------------------------------------------------------------------
// Host-side orchestration
// ---------------------------------------------------------------------------
extern "C" void kernel_launch(void* const* d_in, const int* in_sizes, int n_in,
                              void* d_out, int out_size, void* d_ws, size_t ws_size,
                              hipStream_t stream) {
  (void)in_sizes; (void)n_in; (void)out_size; (void)ws_size;

  const float* x    = (const float*)d_in[0];
  const float* amsk = (const float*)d_in[1];
  const unsigned char* kpm = (const unsigned char*)d_in[2];
  const float* Wq = (const float*)d_in[3];  const float* bq = (const float*)d_in[4];
  const float* Wk = (const float*)d_in[5];  const float* bk = (const float*)d_in[6];
  const float* Wv = (const float*)d_in[7];  const float* bv = (const float*)d_in[8];
  const float* Wg = (const float*)d_in[9];  const float* bg = (const float*)d_in[10];
  const float* Wo = (const float*)d_in[11]; const float* bo = (const float*)d_in[12];

  const size_t M = (size_t)BB * SS;  // 4096 rows
  char* wsp = (char*)d_ws;
  size_t off = 0;
  auto take = [&](size_t bytes) {
    void* p = wsp + off;
    off += (bytes + 255) & ~(size_t)255;
    return p;
  };

  __bf16* xb  = (__bf16*)take(M * DD * sizeof(__bf16));
  __bf16* WtQ = (__bf16*)take((size_t)DD * DD * sizeof(__bf16));
  __bf16* WtK = (__bf16*)take((size_t)DD * DD * sizeof(__bf16));
  __bf16* WtV = (__bf16*)take((size_t)DD * DD * sizeof(__bf16));
  __bf16* WtG = (__bf16*)take((size_t)DD * DD * sizeof(__bf16));
  __bf16* WtO = (__bf16*)take((size_t)DD * DD * sizeof(__bf16));
  __bf16* Qb  = (__bf16*)take(M * DD * sizeof(__bf16));
  __bf16* Kb  = (__bf16*)take(M * DD * sizeof(__bf16));
  __bf16* Vtb = (__bf16*)take(M * DD * sizeof(__bf16));
  __bf16* Gb  = (__bf16*)take(M * DD * sizeof(__bf16));
  __bf16* AOb = (__bf16*)take(M * DD * sizeof(__bf16));
  float*  Mbuf = (float*)take((size_t)BB * HH * SS * sizeof(float));
  float*  Lbuf = (float*)take((size_t)BB * HH * SS * sizeof(float));

  float* out_proj = (float*)d_out;                           // (B,S,D) fp32
  float* avg = out_proj + M * DD;                            // (B,S,S) fp32

  // 1) convert x to bf16
  {
    int n = (int)(M * DD);
    cvt_bf16_kernel<<<(n + 255) / 256, 256, 0, stream>>>(x, xb, n);
  }
  // 2) transpose+convert the 5 weight matrices
  {
    dim3 g(DD / 32, DD / 32), blk(32, 8);
    transpose_cvt_kernel<<<g, blk, 0, stream>>>(Wq, WtQ);
    transpose_cvt_kernel<<<g, blk, 0, stream>>>(Wk, WtK);
    transpose_cvt_kernel<<<g, blk, 0, stream>>>(Wv, WtV);
    transpose_cvt_kernel<<<g, blk, 0, stream>>>(Wg, WtG);
    transpose_cvt_kernel<<<g, blk, 0, stream>>>(Wo, WtO);
  }
  // 3) projections: Q, K, V^T, gate
  {
    dim3 g(M / 128, DD / 64);
    gemm_proj_kernel<<<g, 256, 0, stream>>>(xb, WtQ, bq, Qb,  nullptr, 0);
    gemm_proj_kernel<<<g, 256, 0, stream>>>(xb, WtK, bk, Kb,  nullptr, 1);
    gemm_proj_kernel<<<g, 256, 0, stream>>>(xb, WtV, bv, Vtb, nullptr, 2);
    gemm_proj_kernel<<<g, 256, 0, stream>>>(xb, WtG, bg, Gb,  nullptr, 3);
  }
  // 4) attention pass 1: softmax row stats
  {
    dim3 g(SS / 128, HH, BB);
    attn_pass1_kernel<<<g, 256, 0, stream>>>(Qb, Kb, amsk, kpm, Mbuf, Lbuf);
  }
  // 5) zero avg_attn (atomic accumulation target), then pass 2
  hipMemsetAsync(avg, 0, (size_t)BB * SS * SS * sizeof(float), stream);
  {
    dim3 g(SS / 128, HH, BB);
    attn_pass2_kernel<<<g, 256, 0, stream>>>(Qb, Kb, Vtb, Gb, amsk, kpm,
                                             Mbuf, Lbuf, avg, AOb);
  }
  // 6) output projection (fp32 result straight to d_out)
  {
    dim3 g(M / 128, DD / 64);
    gemm_proj_kernel<<<g, 256, 0, stream>>>(AOb, WtO, bo, nullptr, out_proj, 4);
  }
}
